// HyperNetwork1Layer_60069412602345
// MI455X (gfx1250) — matrix operand order repaired and checked by reading, not compiled, not run on hardware
//
#include <hip/hip_runtime.h>
#include <hip/hip_bf16.h>
#include <stdint.h>

// ---------------------------------------------------------------------------
// HyperNetwork fused via Khatri-Rao GEMM:
//   h1 = relu((Z1 @ W1^T + x@biw^T) * s1 + b1),  Z1[b,(i,m)] = x[b,i]*meta[b,m]
//   (same for layers 2 and 3).  All big GEMMs run on v_wmma_f32_16x16x32_bf16.
// ---------------------------------------------------------------------------

#define BSZ   2048
#define METAD 256
#define HID   128
#define OUTD  64
#define IN1   512   // BASE + META
#define IBLK  2     // i-steps staged per LDS block

typedef __bf16 v16bf __attribute__((ext_vector_type(16)));
typedef __bf16 bf2   __attribute__((ext_vector_type(2)));
typedef float  v8f   __attribute__((ext_vector_type(8)));
typedef unsigned int v8u __attribute__((ext_vector_type(8)));

static __device__ __forceinline__ unsigned pkbf(float lo, float hi) {
  bf2 t;
  t[0] = (__bf16)lo;
  t[1] = (__bf16)hi;
  return __builtin_bit_cast(unsigned, t);
}

// A-matrix (16x32 bf16) K index for VGPR v, this lane (ISA 7.12.2)
static __device__ __forceinline__ int kA(int v, int lane) {
  return ((v < 4) ? (2 * v) : (2 * v + 8)) + ((lane & 16) ? 8 : 0);
}
// Contiguous K index (16h + 2v) for VGPR v, this lane; this code's order for both A and B
static __device__ __forceinline__ int kB(int v, int lane) {
  return 2 * v + ((lane & 16) ? 16 : 0);
}

static __device__ __forceinline__ v8f wmma_bf16x32(const unsigned a[8],
                                                   const unsigned b[8], v8f c) {
  v8u au, bu;
#pragma unroll
  for (int i = 0; i < 8; ++i) { au[i] = a[i]; bu[i] = b[i]; }
  v16bf av = __builtin_bit_cast(v16bf, au);
  v16bf bv = __builtin_bit_cast(v16bf, bu);
  return __builtin_amdgcn_wmma_f32_16x16x32_bf16(false, av, false, bv,
                                                 (short)0, c, false, false);
}

// ---------------------------------------------------------------------------
// x = concat(base, meta)
// ---------------------------------------------------------------------------
__global__ void concat_kernel(const float* __restrict__ base,
                              const float* __restrict__ meta,
                              float* __restrict__ xcat) {
  int t = blockIdx.x * blockDim.x + threadIdx.x;
  if (t >= BSZ * IN1) return;
  int b = t >> 9;       // /512
  int c = t & 511;
  xcat[t] = (c < 256) ? base[b * 256 + c] : meta[b * 256 + (c - 256)];
}

// ---------------------------------------------------------------------------
// Small generic WMMA GEMM: C[M,N] = A[M,K] @ W[N,K]^T (+ bias[n]) (+ addc)
// grid = (M/64, N/16), block = 128 (4 waves, one 16x16 tile each)
// ---------------------------------------------------------------------------
__global__ void gg_kernel(const float* __restrict__ A, int K,
                          const float* __restrict__ W,
                          const float* __restrict__ bias, float addc,
                          float* __restrict__ C, int N) {
  int lane = threadIdx.x & 31;
  int w    = threadIdx.x >> 5;
  int b0   = blockIdx.x * 64 + w * 16;
  int n0   = blockIdx.y * 16;
  int brow = b0 + (lane & 15);
  int nrow = n0 + (lane & 15);

  v8f acc = {};
  int nchunks = K >> 5;
  for (int c = 0; c < nchunks; ++c) {
    unsigned af[8], bfg[8];
#pragma unroll
    for (int v = 0; v < 8; ++v) {
      int ka = c * 32 + kB(v, lane);
      const float* ap = A + (size_t)brow * K + ka;
      af[v] = pkbf(ap[0], ap[1]);
      int kb = c * 32 + kB(v, lane);
      const float* wp = W + (size_t)nrow * K + kb;
      bfg[v] = pkbf(wp[0], wp[1]);
    }
    acc = wmma_bf16x32(af, bfg, acc);
  }
  float bn = (bias != nullptr) ? bias[nrow] : 0.0f;
#pragma unroll
  for (int v = 0; v < 8; ++v) {
    int bb = b0 + v + ((lane & 16) ? 8 : 0);
    C[(size_t)bb * N + n0 + (lane & 15)] = acc[v] + bn + addc;
  }
}

// ---------------------------------------------------------------------------
// Fused hyper-layer:
//   Out[b,n] = act( (sum_{i,m} V[b,i]*meta[b,m]*Wgen[n*I+i, m] + xb[b,n])
//                   * S[b,n] + Bv[b,n] )
// grid = (B/64, N/16), block = 128 (4 waves). LDS double-buffers IBLK i-steps
// of Wgen rows, converted f32 -> bf16 on the fly.
// ---------------------------------------------------------------------------
__global__ void fused_kernel(const float* __restrict__ V, int I,
                             const float* __restrict__ meta,
                             const float* __restrict__ Wgen,
                             const float* __restrict__ xb,
                             const float* __restrict__ S,
                             const float* __restrict__ Bv,
                             float* __restrict__ Out, int N, int do_relu) {
  // [buf][j][n][m/2] packed bf16 pairs : 2*IBLK*16*128 u32 = 32 KB
  __shared__ __align__(16) unsigned wlds[2 * IBLK * 16 * 128];

  int tid  = threadIdx.x;
  int lane = tid & 31;
  int w    = tid >> 5;
  int b0   = blockIdx.x * 64 + w * 16;
  int n0   = blockIdx.y * 16;
  int brow = b0 + (lane & 15);

  // meta bf16 A-fragments, cached once (reused for every i)
  unsigned mf[8][8];
#pragma unroll
  for (int c = 0; c < 8; ++c)
#pragma unroll
    for (int v = 0; v < 8; ++v) {
      int k = c * 32 + kB(v, lane);
      mf[c][v] = pkbf(meta[(size_t)brow * METAD + k],
                      meta[(size_t)brow * METAD + k + 1]);
    }

  v8f acc = {};

  // cooperative stage of Wgen rows (n0..n0+15) x (ib..ib+IBLK-1), f32->bf16
  auto stage = [&](int buf, int ib) {
    int rr = tid >> 2;         // 0..31 : row within block
    int q  = tid & 3;          // quarter of the 256-wide m row
    int n  = rr & 15;
    int j  = rr >> 4;
    const float* src = Wgen + (size_t)((n0 + n) * I + (ib + j)) * METAD + q * 64;
    unsigned* dst = wlds + ((buf * IBLK + j) * 16 + n) * 128 + q * 32;
#pragma unroll
    for (int t = 0; t < 16; ++t) {
      float4 f = ((const float4*)src)[t];
      dst[t * 2 + 0] = pkbf(f.x, f.y);
      dst[t * 2 + 1] = pkbf(f.z, f.w);
    }
  };

  stage(0, 0);
  __syncthreads();

  int p = 0;
  for (int ib = 0; ib < I; ib += IBLK) {
    if (ib + IBLK < I) stage(p ^ 1, ib + IBLK);  // overlap next stage
#pragma unroll
    for (int j = 0; j < IBLK; ++j) {
      int i = ib + j;
      float xs = V[(size_t)brow * I + i];
      bf2 xs2;
      xs2[0] = (__bf16)xs;
      xs2[1] = xs2[0];
      const unsigned* wp =
          wlds + ((p * IBLK + j) * 16 + (lane & 15)) * 128 + ((lane & 16) ? 8 : 0);
#pragma unroll
      for (int c = 0; c < 8; ++c) {
        unsigned af[8], bfg[8];
        uint4 lo = *(const uint4*)(wp + c * 16);
        uint4 hi = *(const uint4*)(wp + c * 16 + 4);
        bfg[0] = lo.x; bfg[1] = lo.y; bfg[2] = lo.z; bfg[3] = lo.w;
        bfg[4] = hi.x; bfg[5] = hi.y; bfg[6] = hi.z; bfg[7] = hi.w;
#pragma unroll
        for (int v = 0; v < 8; ++v) {
          bf2 m2 = __builtin_bit_cast(bf2, mf[c][v]);
          bf2 a2 = m2 * xs2;                    // v_pk_mul_bf16
          af[v]  = __builtin_bit_cast(unsigned, a2);
        }
        acc = wmma_bf16x32(af, bfg, acc);
      }
    }
    __syncthreads();
    p ^= 1;
  }

  int n = n0 + (lane & 15);
#pragma unroll
  for (int v = 0; v < 8; ++v) {
    int bb = b0 + v + ((lane & 16) ? 8 : 0);
    size_t idx = (size_t)bb * N + n;
    float val = acc[v] + xb[idx];
    val = val * S[idx] + Bv[idx];
    if (do_relu) val = fmaxf(val, 0.0f);
    Out[idx] = val;
  }
}

// ---------------------------------------------------------------------------
extern "C" void kernel_launch(void* const* d_in, const int* in_sizes, int n_in,
                              void* d_out, int out_size, void* d_ws,
                              size_t ws_size, hipStream_t stream) {
  (void)in_sizes; (void)n_in; (void)out_size; (void)ws_size;

  const float* meta = (const float*)d_in[0];
  const float* base = (const float*)d_in[1];
  const float* Wi_w = (const float*)d_in[2];
  const float* bi_w = (const float*)d_in[3];   // [128,512] flat
  const float* Wi_b = (const float*)d_in[4];
  const float* bi_b = (const float*)d_in[5];
  const float* Wi_s = (const float*)d_in[6];
  const float* bi_s = (const float*)d_in[7];
  const float* Wh_w = (const float*)d_in[8];
  const float* bh_w = (const float*)d_in[9];   // [128,128] flat
  const float* Wh_b = (const float*)d_in[10];
  const float* bh_b = (const float*)d_in[11];
  const float* Wh_s = (const float*)d_in[12];
  const float* bh_s = (const float*)d_in[13];
  const float* Wl_w = (const float*)d_in[14];
  const float* bl_w = (const float*)d_in[15];  // [64,128] flat
  const float* Wl_b = (const float*)d_in[16];
  const float* bl_b = (const float*)d_in[17];
  const float* Wl_s = (const float*)d_in[18];
  const float* bl_s = (const float*)d_in[19];

  float* ws = (float*)d_ws;
  size_t o = 0;
  float* xcat = ws + o; o += (size_t)BSZ * IN1;
  float* h1   = ws + o; o += (size_t)BSZ * HID;
  float* h2   = ws + o; o += (size_t)BSZ * HID;
  float* s1   = ws + o; o += (size_t)BSZ * HID;
  float* b1   = ws + o; o += (size_t)BSZ * HID;
  float* s2   = ws + o; o += (size_t)BSZ * HID;
  float* b2   = ws + o; o += (size_t)BSZ * HID;
  float* s3   = ws + o; o += (size_t)BSZ * OUTD;
  float* b3   = ws + o; o += (size_t)BSZ * OUTD;
  float* xb1  = ws + o; o += (size_t)BSZ * HID;
  float* xb2  = ws + o; o += (size_t)BSZ * HID;
  float* xb3  = ws + o; o += (size_t)BSZ * OUTD;

  dim3 blk(128);
  dim3 gN128(BSZ / 64, HID / 16);
  dim3 gN64(BSZ / 64, OUTD / 16);

  concat_kernel<<<(BSZ * IN1 + 255) / 256, 256, 0, stream>>>(base, meta, xcat);

  // per-sample scale / bias generators (meta @ W^T + bias [+1])
  gg_kernel<<<gN128, blk, 0, stream>>>(meta, METAD, Wi_s, bi_s, 1.0f, s1, HID);
  gg_kernel<<<gN128, blk, 0, stream>>>(meta, METAD, Wi_b, bi_b, 0.0f, b1, HID);
  gg_kernel<<<gN128, blk, 0, stream>>>(meta, METAD, Wh_s, bh_s, 1.0f, s2, HID);
  gg_kernel<<<gN128, blk, 0, stream>>>(meta, METAD, Wh_b, bh_b, 0.0f, b2, HID);
  gg_kernel<<<gN64,  blk, 0, stream>>>(meta, METAD, Wl_s, bl_s, 1.0f, s3, OUTD);
  gg_kernel<<<gN64,  blk, 0, stream>>>(meta, METAD, Wl_b, bl_b, 0.0f, b3, OUTD);

  // static-bias contributions of generated weight matrices
  gg_kernel<<<gN128, blk, 0, stream>>>(xcat, IN1, bi_w, nullptr, 0.0f, xb1, HID);

  // layer 1 (K = 512*256)
  fused_kernel<<<gN128, blk, 0, stream>>>(xcat, IN1, meta, Wi_w, xb1, s1, b1,
                                          h1, HID, 1);

  gg_kernel<<<gN128, blk, 0, stream>>>(h1, HID, bh_w, nullptr, 0.0f, xb2, HID);

  // layer 2 (K = 128*256)
  fused_kernel<<<gN128, blk, 0, stream>>>(h1, HID, meta, Wh_w, xb2, s2, b2,
                                          h2, HID, 1);

  gg_kernel<<<gN64, blk, 0, stream>>>(h2, HID, bl_w, nullptr, 0.0f, xb3, OUTD);

  // layer 3 (K = 128*256, no relu) -> d_out
  fused_kernel<<<gN64, blk, 0, stream>>>(h2, HID, meta, Wl_w, xb3, s3, b3,
                                         (float*)d_out, OUTD, 0);
}
